// HierarchicalVQVAE_26001732010069
// MI455X (gfx1250) — matrix operand order repaired and checked by reading, not compiled, not run on hardware
//
#include <hip/hip_runtime.h>

typedef __attribute__((ext_vector_type(16))) __bf16 v16bf;
typedef __attribute__((ext_vector_type(8)))  float  v8f;

union Frag { uint4 u[2]; v16bf v; };
union U2S  { uint2 u; unsigned short s[4]; };

__device__ __forceinline__ float bf2f(unsigned short h) {
    return __uint_as_float(((unsigned)h) << 16);
}
__device__ __forceinline__ unsigned short f2bf(float f) {
    unsigned u = __float_as_uint(f);
    unsigned r = u + 0x7FFFu + ((u >> 16) & 1u);   // round-to-nearest-even
    return (unsigned short)(r >> 16);
}

// ---------------------------------------------------------------------------
// Tensor Data Mover staging (CDNA5). This toolchain uses the 6-arg builtin:
//   (uint32x4 g0, int32x8 g1, int32x4 g2, int32x4 g3, int32x8, i32 cpol)
// ---------------------------------------------------------------------------
#if defined(__has_builtin)
#if __has_builtin(__builtin_amdgcn_tensor_load_to_lds) && \
    __has_builtin(__builtin_amdgcn_s_wait_tensorcnt)
#define USE_TDM 1
#endif
#endif
#ifndef USE_TDM
#define USE_TDM 0
#endif

#if USE_TDM
typedef __attribute__((ext_vector_type(4))) unsigned int v4u;
typedef __attribute__((ext_vector_type(8))) int          v8i;
typedef __attribute__((ext_vector_type(4))) int          v4i;

// Load a 2-D bf16 tile [tileRows x 32 elems] (row stride = strideElems) into
// LDS at lds_addr, padded: after each 64B row the DMA skips 16B so the LDS
// layout is 40-ushort (80B) stride -> bank-friendly, 16B-aligned frag reads.
__device__ __forceinline__ void tdm_load_tile(unsigned lds_addr,
                                              const unsigned short* gptr,
                                              int tileRows, int strideElems)
{
    unsigned long long ga = (unsigned long long)(size_t)gptr;
    v4u g0;
    g0.x = 1u;                                        // count=1, user mode
    g0.y = lds_addr;                                  // LDS byte address
    g0.z = (unsigned)(ga & 0xFFFFFFFFu);              // global_addr[31:0]
    g0.w = (unsigned)((ga >> 32) & 0x01FFFFFFu)       // global_addr[56:32]
         | 0x80000000u;                               // type=2 ("image")
    v8i g1;
    g1[0] = (1 << 16)                                 // data_size = 2 bytes
          | (1 << 20)                                 // pad_enable
          | (3 << 22)                                 // pad_interval: 16 DWORDs (64B)
          | (3 << 25);                                // pad_amount:   4 DWORDs (16B)
    g1[1] = (32 & 0xFFFF) << 16;                      // tensor_dim0[15:0] (=32)
    g1[2] = (tileRows & 0xFFFF) << 16;                // tensor_dim0 hi=0 | tensor_dim1 lo
    g1[3] = (32 & 0xFFFF) << 16;                      // tensor_dim1 hi=0 | tile_dim0=32
    g1[4] = tileRows & 0xFFFF;                        // tile_dim1 | tile_dim2=0
    g1[5] = strideElems;                              // tensor_dim0_stride[31:0]
    g1[6] = 0;                                        // stride hi | dim1_stride lo
    g1[7] = 0;
    v4i z4 = {0, 0, 0, 0};                            // groups 2/3 unused (2-D tile)
    v8i z8 = {0, 0, 0, 0, 0, 0, 0, 0};
    __builtin_amdgcn_tensor_load_to_lds(g0, g1, z4, z4, z8, 0);
}
#endif

// ---------------------------------------------------------------------------
// Generic bf16 WMMA GEMM: C[M,N] = epi(A[M,K] @ W[K,N] + bias), W given
// TRANSPOSED as Wt[N,K]. Block = 256 threads (8 waves), tile 128x64, BK=32,
// wave sub-tile 32x32 = 4 x v_wmma_f32_16x16x32_bf16 per k-step.
// Double-buffered LDS; staging via TDM (wave 0 issues 2 descriptors/stage,
// next stage's DMA overlaps current WMMAs, s_wait_tensorcnt <= 2).
// EPI: 0 = relu?+store bf16          (encoder / dec1 activations)
//      1 = store f32 + bf16          (z projection, f32 needed for loss)
//      2 = residual -= v; hsum +=/= v (post projection)
//      3 = store f32 (recon) + fused sum((v - xref)^2) reduction
// ---------------------------------------------------------------------------
template <int EPI, bool RELU>
__global__ __launch_bounds__(256) void gemm_bf16_wmma(
    const unsigned short* __restrict__ A, const unsigned short* __restrict__ Wt,
    const float* __restrict__ bias, int M, int N, int K,
    float* __restrict__ outF, unsigned short* __restrict__ outB,
    unsigned short* __restrict__ resid, unsigned short* __restrict__ hsum,
    int hsumStore, const float* __restrict__ xref, float* __restrict__ accum)
{
    // 80B row stride (32 data ushorts + 8 pad) -> all frag reads ds_load_b128
    __shared__ __attribute__((aligned(16))) unsigned short sA[2][128 * 40];
    __shared__ __attribute__((aligned(16))) unsigned short sB[2][64 * 40];
    __shared__ float sRed[8];

    const int tid  = threadIdx.x;
    const int lane = tid & 31;
    const int wave = tid >> 5;
    const int wm   = wave >> 1;   // 0..3  (rows)
    const int wn   = wave & 1;    // 0..1  (cols)
    const int rowBlock = blockIdx.y * 128;
    const int colBlock = blockIdx.x * 64;
    const int r16 = lane & 15;
    const int kbA = (lane < 16) ? 0 : 8;    // ISA A-fragment K base per half-wave
    const int kbB = (lane < 16) ? 0 : 8;    // ISA B-fragment K base per half-wave

    v8f acc[2][2];
    v8f vz = {};
    acc[0][0] = vz; acc[0][1] = vz; acc[1][0] = vz; acc[1][1] = vz;

    auto stage = [&](int buf, int kk) {
#if USE_TDM
        if (wave == 0) {
            tdm_load_tile((unsigned)(size_t)&sA[buf][0],
                          A + (size_t)rowBlock * K + kk, 128, K);
            tdm_load_tile((unsigned)(size_t)&sB[buf][0],
                          Wt + (size_t)colBlock * K + kk, 64, K);
        }
#else
        int c = tid, r = c >> 2, c8 = (c & 3) * 8;
        *(uint4*)&sA[buf][r * 40 + c8] =
            *(const uint4*)&A[(size_t)(rowBlock + r) * K + kk + c8];
        c = tid + 256; r = c >> 2; c8 = (c & 3) * 8;
        *(uint4*)&sA[buf][r * 40 + c8] =
            *(const uint4*)&A[(size_t)(rowBlock + r) * K + kk + c8];
        int rb = tid >> 2, b8 = (tid & 3) * 8;
        *(uint4*)&sB[buf][rb * 40 + b8] =
            *(const uint4*)&Wt[(size_t)(colBlock + rb) * K + kk + b8];
#endif
    };

    stage(0, 0);
    int buf = 0;
    for (int kk = 0; kk < K; kk += 32) {
        const bool hasNext = (kk + 32) < K;
        if (hasNext) stage(buf ^ 1, kk + 32);     // DMA of next tile overlaps compute
#if USE_TDM
        if (wave == 0) {
            if (hasNext) __builtin_amdgcn_s_wait_tensorcnt(2);  // current stage done
            else         __builtin_amdgcn_s_wait_tensorcnt(0);
        }
#endif
        __syncthreads();

        Frag aF[2], bF[2];
        #pragma unroll
        for (int m = 0; m < 2; ++m) {
            int row = wm * 32 + m * 16 + r16;
            aF[m].u[0] = *(const uint4*)&sA[buf][row * 40 + kbA];       // K=kbA..+7
            aF[m].u[1] = *(const uint4*)&sA[buf][row * 40 + kbA + 16];  // K=kbA+16..+23
        }
        #pragma unroll
        for (int n = 0; n < 2; ++n) {
            int col = wn * 32 + n * 16 + r16;
            bF[n].u[0] = *(const uint4*)&sB[buf][col * 40 + kbB];       // K=kbB..+7
            bF[n].u[1] = *(const uint4*)&sB[buf][col * 40 + kbB + 16];  // K=kbB+16..+23
        }
        #pragma unroll
        for (int m = 0; m < 2; ++m)
            #pragma unroll
            for (int n = 0; n < 2; ++n)
                acc[m][n] = __builtin_amdgcn_wmma_f32_16x16x32_bf16(
                    false, aF[m].v, false, bF[n].v, (short)0, acc[m][n], false, false);
        __syncthreads();
        buf ^= 1;
    }

    // Epilogue. C/D layout: VGPR r, lane l -> M = r + (l<16?0:8), N = l%16.
    float lsum = 0.0f;
    const int mh = (lane < 16) ? 0 : 8;
    #pragma unroll
    for (int m = 0; m < 2; ++m) {
        #pragma unroll
        for (int n = 0; n < 2; ++n) {
            int col = colBlock + wn * 32 + n * 16 + r16;
            float bv = bias ? bias[col] : 0.0f;
            #pragma unroll
            for (int r = 0; r < 8; ++r) {
                int row = rowBlock + wm * 32 + m * 16 + r + mh;
                float v = acc[m][n][r] + bv;
                if (RELU) v = fmaxf(v, 0.0f);
                size_t off = (size_t)row * N + col;
                if constexpr (EPI == 0) {
                    outB[off] = f2bf(v);
                } else if constexpr (EPI == 1) {
                    outF[off] = v;
                    outB[off] = f2bf(v);
                } else if constexpr (EPI == 2) {
                    float rv = bf2f(resid[off]) - v;
                    resid[off] = f2bf(rv);
                    float hv = hsumStore ? v : (bf2f(hsum[off]) + v);
                    hsum[off] = f2bf(hv);
                } else { // EPI == 3: recon + fused MSE vs x
                    outF[off] = v;
                    float d = v - xref[off];
                    lsum += d * d;
                }
            }
        }
    }
    if constexpr (EPI == 3) {
        #pragma unroll
        for (int m = 16; m >= 1; m >>= 1) lsum += __shfl_xor(lsum, m, 32);
        if (lane == 0) sRed[wave] = lsum;
        __syncthreads();
        if (tid == 0) {
            float t = 0.f;
            #pragma unroll
            for (int w = 0; w < 8; ++w) t += sRed[w];
            atomicAdd(&accum[1], t);
        }
    }
}

// ---------------------------------------------------------------------------
// VQ: per wave, one 16x64 z tile (A-fragments in registers), sweep codebook
// in 16-code tiles with 2 WMMAs each; argmin of (||cb||^2 - 2 z.cb), then
// gather q + loss. B-fragments load straight from the row-major bf16 codebook
// (two contiguous 16B global loads per lane per fragment).
// ---------------------------------------------------------------------------
__global__ __launch_bounds__(256) void vq_argmin_gather(
    const unsigned short* __restrict__ zB, const float* __restrict__ zF,
    const unsigned short* __restrict__ cb, const float* __restrict__ cbn,
    int Kcb, float* __restrict__ idsOut, unsigned short* __restrict__ qB,
    float* __restrict__ accum)
{
    __shared__ int   sIdx[128];
    __shared__ float sRed[8];
    const int tid  = threadIdx.x;
    const int lane = tid & 31;
    const int wave = tid >> 5;
    const int rowBlock = blockIdx.x * 128;
    const int rowBase  = rowBlock + wave * 16;
    const int r16 = lane & 15;
    const int kbA = (lane < 16) ? 0 : 8;
    const int kbB = (lane < 16) ? 0 : 8;

    Frag a0, a1;
    {
        size_t rz = (size_t)(rowBase + r16) * 64;
        a0.u[0] = *(const uint4*)&zB[rz + kbA];
        a0.u[1] = *(const uint4*)&zB[rz + kbA + 16];
        a1.u[0] = *(const uint4*)&zB[rz + 32 + kbA];
        a1.u[1] = *(const uint4*)&zB[rz + 32 + kbA + 16];
    }

    float best[8]; int bidx[8];
    #pragma unroll
    for (int r = 0; r < 8; ++r) { best[r] = 3.4e38f; bidx[r] = 0; }

    for (int ct = 0; ct < (Kcb >> 4); ++ct) {
        int code = ct * 16 + r16;
        size_t rc = (size_t)code * 64;
        Frag b0, b1;
        b0.u[0] = *(const uint4*)&cb[rc + kbB];
        b0.u[1] = *(const uint4*)&cb[rc + kbB + 16];
        b1.u[0] = *(const uint4*)&cb[rc + 32 + kbB];
        b1.u[1] = *(const uint4*)&cb[rc + 32 + kbB + 16];
        v8f s = {};
        s = __builtin_amdgcn_wmma_f32_16x16x32_bf16(false, a0.v, false, b0.v,
                                                    (short)0, s, false, false);
        s = __builtin_amdgcn_wmma_f32_16x16x32_bf16(false, a1.v, false, b1.v,
                                                    (short)0, s, false, false);
        float cn = cbn[code];
        #pragma unroll
        for (int r = 0; r < 8; ++r) {
            float val = cn - 2.0f * s[r];   // N index of s[r] at this lane == code
            if (val < best[r]) { best[r] = val; bidx[r] = code; }
        }
    }
    #pragma unroll
    for (int r = 0; r < 8; ++r) {           // reduce within each 16-lane half
        #pragma unroll
        for (int m = 1; m < 16; m <<= 1) {
            float ob = __shfl_xor(best[r], m, 32);
            int   oi = __shfl_xor(bidx[r], m, 32);
            if (ob < best[r] || (ob == best[r] && oi < bidx[r])) {
                best[r] = ob; bidx[r] = oi;
            }
        }
    }
    if (r16 == 0) {                          // lane 0 -> rows 0..7, lane 16 -> 8..15
        int half = (lane >> 4) * 8;
        #pragma unroll
        for (int r = 0; r < 8; ++r) {
            sIdx[wave * 16 + half + r] = bidx[r];
            idsOut[rowBase + half + r] = (float)bidx[r];
        }
    }
    __syncthreads();

    float lsum = 0.0f;
    for (int t = tid; t < 128 * 64; t += 256) {
        int rl = t >> 6, d = t & 63;
        int code = sIdx[rl];
        unsigned short qh = cb[(size_t)code * 64 + d];
        size_t off = (size_t)(rowBlock + rl) * 64 + d;
        qB[off] = qh;
        float diff = bf2f(qh) - zF[off];
        lsum += diff * diff;
    }
    #pragma unroll
    for (int m = 16; m >= 1; m >>= 1) lsum += __shfl_xor(lsum, m, 32);
    if (lane == 0) sRed[wave] = lsum;
    __syncthreads();
    if (tid == 0) {
        float t = 0.f;
        #pragma unroll
        for (int w = 0; w < 8; ++w) t += sRed[w];
        atomicAdd(&accum[0], t);
    }
}

// ---------------------------------------------------------------------------
// Small helper kernels
// ---------------------------------------------------------------------------
__global__ void cvt_f32_bf16(const float* __restrict__ in,
                             unsigned short* __restrict__ out, int n)
{
    int i = (blockIdx.x * 256 + threadIdx.x) * 4;
    if (i + 3 < n) {
        float4 v = *(const float4*)&in[i];
        U2S t;
        t.s[0] = f2bf(v.x); t.s[1] = f2bf(v.y);
        t.s[2] = f2bf(v.z); t.s[3] = f2bf(v.w);
        *(uint2*)&out[i] = t.u;
    }
}

// W[K,N] f32 -> Wt[N,K] bf16 (weights are small; coalesced reads)
__global__ void cvt_transpose(const float* __restrict__ in,
                              unsigned short* __restrict__ out, int K, int N)
{
    int idx = blockIdx.x * 256 + threadIdx.x;
    if (idx < K * N) {
        int k = idx / N, n = idx % N;
        out[(size_t)n * K + k] = f2bf(in[idx]);
    }
}

__global__ void cb_prep(const float* __restrict__ cb,
                        unsigned short* __restrict__ cbB,
                        float* __restrict__ cbn, int Kcb)
{
    int k = blockIdx.x * 256 + threadIdx.x;
    if (k >= Kcb) return;
    float s = 0.0f;
    #pragma unroll 4
    for (int d = 0; d < 64; ++d) {
        float v = cb[(size_t)k * 64 + d];
        s += v * v;
        cbB[(size_t)k * 64 + d] = f2bf(v);
    }
    cbn[k] = s;
}

__global__ void zero_accum(float* a)
{
    if (threadIdx.x < 2) a[threadIdx.x] = 0.0f;
}

__global__ void finalize_loss(const float* __restrict__ accum,
                              float* __restrict__ out)
{
    if (threadIdx.x == 0) {
        float vq    = 1.25f * accum[0] / 4194304.0f;     // B*D_Z = 65536*64
        float recon = accum[1] / 50331648.0f;            // B*D_IN = 65536*768
        out[0] = vq + recon;
    }
}

// ---------------------------------------------------------------------------
extern "C" void kernel_launch(void* const* d_in, const int* in_sizes, int n_in,
                              void* d_out, int out_size, void* d_ws, size_t ws_size,
                              hipStream_t stream)
{
    constexpr int B = 65536, D_IN = 768, D_H = 512, D_Z = 64;
    constexpr int KCB0 = 1024, KCB1 = 512, KCB2 = 256;

    const float* x     = (const float*)d_in[0];
    const float* encW1 = (const float*)d_in[1];
    const float* encB1 = (const float*)d_in[2];
    const float* encW2 = (const float*)d_in[3];
    const float* encB2 = (const float*)d_in[4];
    const float* preW  = (const float*)d_in[5];
    const float* preB  = (const float*)d_in[6];
    const float* cb0   = (const float*)d_in[7];
    const float* cb1   = (const float*)d_in[8];
    const float* cb2   = (const float*)d_in[9];
    const float* postW = (const float*)d_in[10];
    const float* postB = (const float*)d_in[11];
    const float* decW1 = (const float*)d_in[12];
    const float* decB1 = (const float*)d_in[13];
    const float* decW2 = (const float*)d_in[14];
    const float* decB2 = (const float*)d_in[15];

    char* ws = (char*)d_ws;
    size_t o = 256;
    auto alloc = [&](size_t bytes) -> char* {
        char* p = ws + o;
        o += (bytes + 255) & ~(size_t)255;
        return p;
    };
    float*          accum = (float*)ws;                                  // [2]
    unsigned short* xB    = (unsigned short*)alloc((size_t)B * D_IN * 2);
    unsigned short* wE1   = (unsigned short*)alloc((size_t)D_IN * D_H * 2);  // [512,768]
    unsigned short* wE2   = (unsigned short*)alloc((size_t)D_H * D_H * 2);   // [512,512]
    unsigned short* wPre  = (unsigned short*)alloc((size_t)3 * D_Z * D_H * 2); // [64,512] x3
    unsigned short* wPost = (unsigned short*)alloc((size_t)3 * D_H * D_Z * 2); // [512,64] x3
    unsigned short* wD1   = (unsigned short*)alloc((size_t)D_H * D_H * 2);   // [512,512]
    unsigned short* wD2   = (unsigned short*)alloc((size_t)D_IN * D_H * 2);  // [768,512]
    unsigned short* cbB   = (unsigned short*)alloc((size_t)(KCB0 + KCB1 + KCB2) * D_Z * 2);
    float*          cbN   = (float*)alloc((size_t)(KCB0 + KCB1 + KCB2) * 4);
    unsigned short* h1B   = (unsigned short*)alloc((size_t)B * D_H * 2);  // h1, later hd
    unsigned short* resB  = (unsigned short*)alloc((size_t)B * D_H * 2);  // residual
    unsigned short* hsB   = (unsigned short*)alloc((size_t)B * D_H * 2);  // h_sum
    float*          zF    = (float*)alloc((size_t)B * D_Z * 4);
    unsigned short* zB    = (unsigned short*)alloc((size_t)B * D_Z * 2);
    unsigned short* qB    = (unsigned short*)alloc((size_t)B * D_Z * 2);

    float* reconOut = (float*)d_out;
    float* idsOut   = reconOut + (size_t)B * D_IN;          // [3,B] as f32
    float* lossOut  = idsOut + (size_t)3 * B;               // [1]

    const dim3 blk(256);
    zero_accum<<<1, 32, 0, stream>>>(accum);

    // x -> bf16 (dominant prep traffic: ~300MB @ 23.3 TB/s ~ 13us)
    cvt_f32_bf16<<<(B * D_IN / 4 + 255) / 256, blk, 0, stream>>>(x, xB, B * D_IN);
    // Weights -> transposed bf16 (tiny)
    auto cvtT = [&](const float* src, unsigned short* dst, int K, int N) {
        cvt_transpose<<<(K * N + 255) / 256, blk, 0, stream>>>(src, dst, K, N);
    };
    cvtT(encW1, wE1, D_IN, D_H);
    cvtT(encW2, wE2, D_H, D_H);
    for (int i = 0; i < 3; ++i) {
        cvtT(preW  + (size_t)i * D_H * D_Z, wPre  + (size_t)i * D_Z * D_H, D_H, D_Z);
        cvtT(postW + (size_t)i * D_Z * D_H, wPost + (size_t)i * D_H * D_Z, D_Z, D_H);
    }
    cvtT(decW1, wD1, D_H, D_H);
    cvtT(decW2, wD2, D_H, D_IN);
    cb_prep<<<(KCB0 + 255) / 256, blk, 0, stream>>>(cb0, cbB, cbN, KCB0);
    cb_prep<<<(KCB1 + 255) / 256, blk, 0, stream>>>(cb1, cbB + (size_t)KCB0 * D_Z, cbN + KCB0, KCB1);
    cb_prep<<<(KCB2 + 255) / 256, blk, 0, stream>>>(cb2, cbB + (size_t)(KCB0 + KCB1) * D_Z, cbN + KCB0 + KCB1, KCB2);

    auto G = [](int M, int N) { return dim3(N / 64, M / 128); };

    // Encoder
    gemm_bf16_wmma<0, true><<<G(B, D_H), blk, 0, stream>>>(
        xB, wE1, encB1, B, D_H, D_IN, nullptr, h1B, nullptr, nullptr, 0, nullptr, nullptr);
    gemm_bf16_wmma<0, true><<<G(B, D_H), blk, 0, stream>>>(
        h1B, wE2, encB2, B, D_H, D_H, nullptr, resB, nullptr, nullptr, 0, nullptr, nullptr);

    // VQ levels
    const int kcb[3] = {KCB0, KCB1, KCB2};
    const size_t cbOff[3] = {0, (size_t)KCB0, (size_t)(KCB0 + KCB1)};
    for (int i = 0; i < 3; ++i) {
        gemm_bf16_wmma<1, false><<<G(B, D_Z), blk, 0, stream>>>(
            resB, wPre + (size_t)i * D_Z * D_H, preB + (size_t)i * D_Z,
            B, D_Z, D_H, zF, zB, nullptr, nullptr, 0, nullptr, nullptr);
        vq_argmin_gather<<<B / 128, blk, 0, stream>>>(
            zB, zF, cbB + cbOff[i] * D_Z, cbN + cbOff[i], kcb[i],
            idsOut + (size_t)i * B, qB, accum);
        gemm_bf16_wmma<2, false><<<G(B, D_H), blk, 0, stream>>>(
            qB, wPost + (size_t)i * D_H * D_Z, postB + (size_t)i * D_H,
            B, D_H, D_Z, nullptr, nullptr, resB, hsB, (i == 0) ? 1 : 0, nullptr, nullptr);
    }

    // Decoder (dec2 fuses recon store + MSE reduction vs x)
    gemm_bf16_wmma<0, true><<<G(B, D_H), blk, 0, stream>>>(
        hsB, wD1, decB1, B, D_H, D_H, nullptr, h1B, nullptr, nullptr, 0, nullptr, nullptr);
    gemm_bf16_wmma<3, false><<<G(B, D_IN), blk, 0, stream>>>(
        h1B, wD2, decB2, B, D_IN, D_H, reconOut, nullptr, nullptr, nullptr, 0, x, accum);

    finalize_loss<<<1, 32, 0, stream>>>(accum, lossOut);
}
